// GatedDeltaNet_17205638988278
// MI455X (gfx1250) — compile-verified, matrix operand order checked by reading, not run on hardware
//
#include <hip/hip_runtime.h>
#include <cstddef>

#define SEQ 4096
#define HS 2048
#define NK 16
#define NV 32
#define DK 128
#define DV 128
#define CHUNK 64
#define NCHUNK (SEQ / CHUNK)
#define KEY_DIM (NK * DK)                 // 2048
#define VAL_DIM (NV * DV)                 // 4096
#define CONV_DIM (2 * KEY_DIM + VAL_DIM)  // 8192

typedef __attribute__((ext_vector_type(16))) __bf16 v16bf;
typedef __attribute__((ext_vector_type(4))) __bf16 v4bf;
typedef __attribute__((ext_vector_type(8))) float v8f;

// ---------------------------------------------------------------------------
// CDNA5 V_WMMA_F32_16X16X32_BF16 helpers (wave32).
// 16-bit A 16x32 layout (ISA 7.12.2): lane L<16 = row L, K {0..7,16..23};
// lane L+16 = row L, K {8..15,24..31}.  A B operand stored TRANSPOSED
// (row = n, ld = K extent) uses identical indexing, so every fragment read
// below is two contiguous 16B runs -> ds_load_b128.
// ---------------------------------------------------------------------------
__device__ __forceinline__ int frag_kbase(int v, int hi) {
  return ((v & 3) << 1) + (hi ? 8 : 0) + ((v >> 2) << 4);
}

__device__ __forceinline__ v8f v8f_zero() {
  v8f z = {0.f, 0.f, 0.f, 0.f, 0.f, 0.f, 0.f, 0.f};
  return z;
}

__device__ __forceinline__ v8f wmma_bf16(v16bf a, v16bf b, v8f c) {
  return __builtin_amdgcn_wmma_f32_16x16x32_bf16(false, a, false, b, (short)0, c,
                                                 false, false);
}

__device__ __forceinline__ v16bf frag_bf16(const __bf16* base, int ld, int row0,
                                           int k0, int lane) {
  int l = lane & 15, hi = lane >> 4;
  const __bf16* rp = base + (row0 + l) * ld + k0;
  v16bf a;
#pragma unroll
  for (int v = 0; v < 8; ++v) {
    int kb = frag_kbase(v, hi);
    a[2 * v] = rp[kb];
    a[2 * v + 1] = rp[kb + 1];
  }
  return a;
}

__device__ __forceinline__ v16bf frag_f32(const float* base, int ld, int row0,
                                          int k0, int lane) {
  int l = lane & 15, hi = lane >> 4;
  const float* rp = base + (row0 + l) * ld + k0;
  v16bf a;
#pragma unroll
  for (int v = 0; v < 8; ++v) {
    int kb = frag_kbase(v, hi);
    a[2 * v] = (__bf16)rp[kb];
    a[2 * v + 1] = (__bf16)rp[kb + 1];
  }
  return a;
}

__device__ __forceinline__ void store_tile(float* base, int ld, int row0, int col0,
                                           int lane, v8f acc, float sgn) {
  int n = lane & 15, hi = lane >> 4;
#pragma unroll
  for (int r = 0; r < 8; ++r)
    base[(size_t)(row0 + r + hi * 8) * ld + col0 + n] = sgn * acc[r];
}

__device__ __forceinline__ void store_tile_bf16(__bf16* base, int ld, int row0,
                                                int col0, int lane, v8f acc,
                                                float sgn) {
  int n = lane & 15, hi = lane >> 4;
#pragma unroll
  for (int r = 0; r < 8; ++r)
    base[(row0 + r + hi * 8) * ld + col0 + n] = (__bf16)(sgn * acc[r]);
}

__device__ __forceinline__ v8f load_tile(const float* base, int ld, int row0,
                                         int col0, int lane) {
  int n = lane & 15, hi = lane >> 4;
  v8f c;
#pragma unroll
  for (int r = 0; r < 8; ++r)
    c[r] = base[(size_t)(row0 + r + hi * 8) * ld + col0 + n];
  return c;
}

__device__ __forceinline__ v4bf pack4(float a, float b, float c, float d) {
  v4bf p;
  p[0] = (__bf16)a;
  p[1] = (__bf16)b;
  p[2] = (__bf16)c;
  p[3] = (__bf16)d;
  return p;
}

// ---------------------------------------------------------------------------
// Kernel 1: WMMA GEMM, C[M,N] = A[M,K] x B[K,N], f32 in/out, bf16 compute.
// Double-buffered LDS; register-staged pipeline: global_load_b128 next tile,
// WMMA current tile, cvt + ds_store_b64 next tile, one barrier / iteration.
// ---------------------------------------------------------------------------
#define GBM 128
#define GBN 128
#define GBK 32

__global__ __launch_bounds__(256) void gemm_f32_bf16wmma(
    const float* __restrict__ A, const float* __restrict__ B,
    float* __restrict__ C, int M, int N, int K) {
  __shared__ alignas(16) __bf16 As[2][GBM * GBK];   // 2 x 8 KB  [m][k]
  __shared__ alignas(16) __bf16 Bst[2][GBN * GBK];  // 2 x 8 KB  [n][k] (B^T)
  const int tid = threadIdx.x;
  const int lane = tid & 31;
  const int w = tid >> 5;
  const int m0 = blockIdx.y * GBM;
  const int n0 = blockIdx.x * GBN;
  const int wr = (w >> 2) * 64;  // 4x2 tiles of 16x16 per wave
  const int wc = (w & 3) * 32;

  v8f acc[4][2];
#pragma unroll
  for (int i = 0; i < 4; ++i) {
    acc[i][0] = v8f_zero();
    acc[i][1] = v8f_zero();
  }

  // ---- prologue: stage k0 = 0 into buffer 0 -----------------------------
#pragma unroll
  for (int t = 0; t < 4; ++t) {
    int j = tid + 256 * t;
    int row = j >> 3, c4 = j & 7;
    float4 av = *(const float4*)&A[(size_t)(m0 + row) * K + c4 * 4];
    *(v4bf*)&As[0][row * GBK + c4 * 4] = pack4(av.x, av.y, av.z, av.w);
    int n = j & 127, rr = (j >> 7) * 4;
    v4bf pb;
#pragma unroll
    for (int u = 0; u < 4; ++u)
      pb[u] = (__bf16)B[(size_t)(rr + u) * N + n0 + n];
    *(v4bf*)&Bst[0][n * GBK + rr] = pb;
  }
  __syncthreads();

  for (int k0 = 0; k0 < K; k0 += GBK) {
    const int buf = (k0 >> 5) & 1;
    const int nxt = buf ^ 1;
    const bool more = (k0 + GBK) < K;
    const int kn = k0 + GBK;

    float4 areg[4];
    float breg[4][4];
    if (more) {
#pragma unroll
      for (int t = 0; t < 4; ++t) {
        int j = tid + 256 * t;
        int row = j >> 3, c4 = j & 7;
        areg[t] = *(const float4*)&A[(size_t)(m0 + row) * K + kn + c4 * 4];
        int n = j & 127, rr = (j >> 7) * 4;
#pragma unroll
        for (int u = 0; u < 4; ++u)
          breg[t][u] = B[(size_t)(kn + rr + u) * N + n0 + n];
      }
      __builtin_prefetch(&B[(size_t)(kn + GBK + (tid >> 3)) * N + n0 + (tid & 7) * 16], 0, 0);
    }

    // ---- compute on current buffer --------------------------------------
    v16bf bf0 = frag_bf16(&Bst[buf][0], GBK, wc, 0, lane);
    v16bf bf1 = frag_bf16(&Bst[buf][0], GBK, wc + 16, 0, lane);
#pragma unroll
    for (int tr = 0; tr < 4; ++tr) {
      v16bf af = frag_bf16(&As[buf][0], GBK, wr + tr * 16, 0, lane);
      acc[tr][0] = wmma_bf16(af, bf0, acc[tr][0]);
      acc[tr][1] = wmma_bf16(af, bf1, acc[tr][1]);
    }

    // ---- store next tile ------------------------------------------------
    if (more) {
#pragma unroll
      for (int t = 0; t < 4; ++t) {
        int j = tid + 256 * t;
        int row = j >> 3, c4 = j & 7;
        *(v4bf*)&As[nxt][row * GBK + c4 * 4] =
            pack4(areg[t].x, areg[t].y, areg[t].z, areg[t].w);
        int n = j & 127, rr = (j >> 7) * 4;
        *(v4bf*)&Bst[nxt][n * GBK + rr] =
            pack4(breg[t][0], breg[t][1], breg[t][2], breg[t][3]);
      }
    }
    __syncthreads();
  }

  float* Cb = C + (size_t)m0 * N + n0;
#pragma unroll
  for (int tr = 0; tr < 4; ++tr) {
    store_tile(Cb, N, wr + tr * 16, wc, lane, acc[tr][0], 1.f);
    store_tile(Cb, N, wr + tr * 16, wc + 16, lane, acc[tr][1], 1.f);
  }
}

// ---------------------------------------------------------------------------
// Kernel 2: beta = sigmoid(h@W_b), g = -exp(A_log)*softplus(h@W_a + dt_bias)
// ---------------------------------------------------------------------------
__global__ __launch_bounds__(64) void gb_kernel(const float* __restrict__ hidden,
                                                const float* __restrict__ Wb,
                                                const float* __restrict__ Wa,
                                                const float* __restrict__ dtb,
                                                const float* __restrict__ Alog,
                                                float* __restrict__ bArr,
                                                float* __restrict__ gArr) {
  __shared__ alignas(16) float row[HS];
  const int s = blockIdx.x;
  const int t = threadIdx.x;
  for (int i = t; i < HS / 4; i += 64)
    ((float4*)row)[i] = ((const float4*)(hidden + (size_t)s * HS))[i];
  __syncthreads();
  if (t < NV) {
    float acc = 0.f;
    for (int i = 0; i < HS / 4; ++i) {
      float4 rv = ((const float4*)row)[i];
      const float* wp = Wb + (size_t)(4 * i) * NV + t;
      acc += rv.x * wp[0] + rv.y * wp[NV] + rv.z * wp[2 * NV] + rv.w * wp[3 * NV];
    }
    bArr[(size_t)s * NV + t] = 1.f / (1.f + __expf(-acc));
  } else {
    int c = t - NV;
    float acc = 0.f;
    for (int i = 0; i < HS / 4; ++i) {
      float4 rv = ((const float4*)row)[i];
      const float* wp = Wa + (size_t)(4 * i) * NV + c;
      acc += rv.x * wp[0] + rv.y * wp[NV] + rv.z * wp[2 * NV] + rv.w * wp[3 * NV];
    }
    float x = acc + dtb[c];
    float sp = (x > 20.f) ? x : log1pf(__expf(x));
    gArr[(size_t)s * NV + c] = -__expf(Alog[c]) * sp;
  }
}

// ---------------------------------------------------------------------------
// Kernel 3: depthwise causal conv (K=4) + SiLU, 4 channels per thread.
// ---------------------------------------------------------------------------
__device__ __forceinline__ float silu1(float x) { return x / (1.f + __expf(-x)); }

__global__ __launch_bounds__(256) void conv_silu(const float* __restrict__ x,
                                                 const float* __restrict__ cw,
                                                 float* __restrict__ out) {
  const int C4 = CONV_DIM / 4;
  const size_t total4 = (size_t)SEQ * C4;
  const float4* xp = (const float4*)x;
  const float4* wp = (const float4*)cw;  // wp[c] = {w[c][0..3]}
  float4* op = (float4*)out;
  const float4 zero = {0.f, 0.f, 0.f, 0.f};
  for (size_t j = (size_t)blockIdx.x * blockDim.x + threadIdx.x; j < total4;
       j += (size_t)gridDim.x * blockDim.x) {
    int c4 = (int)(j % C4);
    size_t s = j / C4;
    float4 x0 = xp[j];
    float4 x1 = (s >= 1) ? xp[j - C4] : zero;
    float4 x2 = (s >= 2) ? xp[j - 2 * (size_t)C4] : zero;
    float4 x3 = (s >= 3) ? xp[j - 3 * (size_t)C4] : zero;
    float4 w0 = wp[c4 * 4 + 0], w1 = wp[c4 * 4 + 1];
    float4 w2 = wp[c4 * 4 + 2], w3 = wp[c4 * 4 + 3];
    float4 o;
    o.x = silu1(x0.x * w0.w + x1.x * w0.z + x2.x * w0.y + x3.x * w0.x);
    o.y = silu1(x0.y * w1.w + x1.y * w1.z + x2.y * w1.y + x3.y * w1.x);
    o.z = silu1(x0.z * w2.w + x1.z * w2.z + x2.z * w2.y + x3.z * w2.x);
    o.w = silu1(x0.w * w3.w + x1.w * w3.z + x2.w * w3.y + x3.w * w3.x);
    op[j] = o;
  }
}

// ---------------------------------------------------------------------------
// Kernel 4: chunked gated delta rule core. One workgroup (8 waves) per head.
// fp32 state/solve; bf16 operand mirrors; LDS ~313 KB (CDNA5 WGP: 320 KB).
// ---------------------------------------------------------------------------
__global__ __launch_bounds__(256) void delta_core(const float* __restrict__ qkv,
                                                  const float* __restrict__ gArr,
                                                  const float* __restrict__ bArr,
                                                  float* __restrict__ coreo) {
  __shared__ alignas(16) float S[DK * DV];           // 64 KB f32 state
  __shared__ alignas(16) __bf16 St[DV * DK];         // 32 KB S^T (B operand)
  __shared__ alignas(16) float qc[CHUNK * DK];       // 32 KB q -> q*e^g
  __shared__ alignas(16) float kc[CHUNK * DK];       // 32 KB k (l2norm)
  __shared__ alignas(16) __bf16 kb16[CHUNK * DK];    // 16 KB k bf16 (A / B^T)
  __shared__ alignas(16) float bufV[CHUNK * DV];     // 32 KB v*beta -> v_new
  __shared__ alignas(16) __bf16 Vt[DV * CHUNK];      // 16 KB bufV^T (B)
  __shared__ alignas(16) __bf16 Xt[DK * CHUNK];      // 16 KB (k b e^g)^T (B)
  __shared__ alignas(16) __bf16 Wb[CHUNK * DK];      // 16 KB -w (A)
  __shared__ alignas(16) __bf16 Kd[DK * CHUNK];      // 16 KB k^T e^(gl-g) (A)
  __shared__ alignas(16) float Amat[CHUNK * CHUNK];  // 16 KB KK->A->QK->attn
  __shared__ alignas(16) float Tm[CHUNK * CHUNK];    // 16 KB (I-A)^-1
  __shared__ float gc[CHUNK], eg[CHUNK], ef[CHUNK], betas[CHUNK];
  __shared__ float qs[CHUNK], ks[CHUNK];
  __shared__ float glast;

  const int h = blockIdx.x;
  const int tid = threadIdx.x;
  const int lane = tid & 31;
  const int w = tid >> 5;
  const int qoff = (h >> 1) * DK;  // GQA: 2 v-heads per qk-head
  const int koff = KEY_DIM + (h >> 1) * DK;
  const int voff = 2 * KEY_DIM + h * DV;

  for (int i = tid; i < DK * DV; i += 256) {
    S[i] = 0.f;
    St[i] = (__bf16)0.f;
  }
  __syncthreads();

  for (int c = 0; c < NCHUNK; ++c) {
    const int s0 = c * CHUNK;
    // ---- load raw q,k,v chunk (float4 / b128) ---------------------------
#pragma unroll
    for (int t = 0; t < 8; ++t) {
      int j = tid + 256 * t;  // float4 index over 64x128
      int r = j >> 5, d4 = j & 31;
      const float* rowp = qkv + (size_t)(s0 + r) * CONV_DIM;
      ((float4*)qc)[j] = *(const float4*)(rowp + qoff + d4 * 4);
      ((float4*)kc)[j] = *(const float4*)(rowp + koff + d4 * 4);
      ((float4*)bufV)[j] = *(const float4*)(rowp + voff + d4 * 4);
    }
    __syncthreads();
    // ---- per-row stats + gate cumsum ------------------------------------
    if (tid < CHUNK) {
      int r = tid;
      const float4* qr = (const float4*)(qc + r * DK);
      const float4* kr = (const float4*)(kc + r * DK);
      float sq = 0.f, sk = 0.f;
      for (int d = 0; d < DK / 4; ++d) {
        float4 a = qr[d];
        sq += a.x * a.x + a.y * a.y + a.z * a.z + a.w * a.w;
        float4 b = kr[d];
        sk += b.x * b.x + b.y * b.y + b.z * b.z + b.w * b.w;
      }
      qs[r] = rsqrtf(sq + 1e-6f) * 0.08838834764831845f;  // * dk^-0.5
      ks[r] = rsqrtf(sk + 1e-6f);
      betas[r] = bArr[(size_t)(s0 + r) * NV + h];
    }
    if (tid == 0) {
      float a = 0.f;
      for (int r = 0; r < CHUNK; ++r) {
        a += gArr[(size_t)(s0 + r) * NV + h];
        gc[r] = a;
      }
      glast = a;
    }
    __syncthreads();
    if (tid < CHUNK) {
      eg[tid] = __expf(gc[tid]);
      ef[tid] = __expf(glast - gc[tid]);
    }
    __syncthreads();
    // ---- normalize + v*beta (float4) ------------------------------------
    for (int j = tid; j < CHUNK * DK / 4; j += 256) {
      int r = j >> 5;
      float4 kv = ((float4*)kc)[j];
      float s1 = ks[r];
      kv.x *= s1; kv.y *= s1; kv.z *= s1; kv.w *= s1;
      ((float4*)kc)[j] = kv;
      float4 qv = ((float4*)qc)[j];
      float s2 = qs[r];
      qv.x *= s2; qv.y *= s2; qv.z *= s2; qv.w *= s2;
      ((float4*)qc)[j] = qv;
      float4 vv = ((float4*)bufV)[j];
      float s3 = betas[r];
      vv.x *= s3; vv.y *= s3; vv.z *= s3; vv.w *= s3;
      ((float4*)bufV)[j] = vv;
    }
    __syncthreads();
    // ---- build bf16 operand mirrors -------------------------------------
    for (int j = tid; j < CHUNK * DK / 4; j += 256) {
      int r = j >> 5, d0 = (j & 31) * 4;
      float4 kv = ((const float4*)kc)[j];
      float4 vv = ((const float4*)bufV)[j];
      ((v4bf*)kb16)[j] = pack4(kv.x, kv.y, kv.z, kv.w);
      float be = betas[r] * eg[r], fe = ef[r];
      Xt[(d0 + 0) * CHUNK + r] = (__bf16)(kv.x * be);
      Xt[(d0 + 1) * CHUNK + r] = (__bf16)(kv.y * be);
      Xt[(d0 + 2) * CHUNK + r] = (__bf16)(kv.z * be);
      Xt[(d0 + 3) * CHUNK + r] = (__bf16)(kv.w * be);
      Kd[(d0 + 0) * CHUNK + r] = (__bf16)(kv.x * fe);
      Kd[(d0 + 1) * CHUNK + r] = (__bf16)(kv.y * fe);
      Kd[(d0 + 2) * CHUNK + r] = (__bf16)(kv.z * fe);
      Kd[(d0 + 3) * CHUNK + r] = (__bf16)(kv.w * fe);
      Vt[(d0 + 0) * CHUNK + r] = (__bf16)vv.x;
      Vt[(d0 + 1) * CHUNK + r] = (__bf16)vv.y;
      Vt[(d0 + 2) * CHUNK + r] = (__bf16)vv.z;
      Vt[(d0 + 3) * CHUNK + r] = (__bf16)vv.w;
    }
    __syncthreads();
    // ---- (a) KK = k @ k^T -> Amat (64x64, K=128) ------------------------
    {
      int tr = w >> 1, tc0 = (w & 1) * 2;
      v8f a0 = v8f_zero(), a1 = v8f_zero();
      for (int kk = 0; kk < DK; kk += 32) {
        v16bf af = frag_bf16(kb16, DK, tr * 16, kk, lane);
        v16bf b0 = frag_bf16(kb16, DK, tc0 * 16, kk, lane);
        v16bf b1 = frag_bf16(kb16, DK, (tc0 + 1) * 16, kk, lane);
        a0 = wmma_bf16(af, b0, a0);
        a1 = wmma_bf16(af, b1, a1);
      }
      store_tile(Amat, CHUNK, tr * 16, tc0 * 16, lane, a0, 1.f);
      store_tile(Amat, CHUNK, tr * 16, (tc0 + 1) * 16, lane, a1, 1.f);
    }
    __syncthreads();
    // ---- A = -beta_i*KK*exp(g_i-g_j) strictly lower; T = I --------------
    for (int i = tid; i < CHUNK * CHUNK; i += 256) {
      int r = i >> 6, j = i & 63;
      Amat[i] = (j < r) ? -betas[r] * Amat[i] * __expf(gc[r] - gc[j]) : 0.f;
      Tm[i] = (r == j) ? 1.f : 0.f;
    }
    __syncthreads();
    // ---- T = (I - A)^-1 forward substitution (thread = column) ----------
    if (tid < CHUNK) {
      int col = tid;
      for (int i = col + 1; i < CHUNK; ++i) {
        float s = 0.f;
        for (int j = col; j < i; ++j) s += Amat[i * CHUNK + j] * Tm[j * CHUNK + col];
        Tm[i * CHUNK + col] = s;
      }
    }
    __syncthreads();
    // ---- (c) Wb = -(T @ (k*beta*e^g))  (64x128, K=64) -------------------
    {
      int tr = w >> 1, tc0 = (w & 1) * 4;
      v8f a[4];
#pragma unroll
      for (int t = 0; t < 4; ++t) a[t] = v8f_zero();
      for (int kk = 0; kk < CHUNK; kk += 32) {
        v16bf af = frag_f32(Tm, CHUNK, tr * 16, kk, lane);
#pragma unroll
        for (int t = 0; t < 4; ++t) {
          v16bf b = frag_bf16(Xt, CHUNK, (tc0 + t) * 16, kk, lane);
          a[t] = wmma_bf16(af, b, a[t]);
        }
      }
#pragma unroll
      for (int t = 0; t < 4; ++t)
        store_tile_bf16(Wb, DK, tr * 16, (tc0 + t) * 16, lane, a[t], -1.f);
    }
    __syncthreads();
    // ---- (b) QK = q @ k^T -> Amat ---------------------------------------
    {
      int tr = w >> 1, tc0 = (w & 1) * 2;
      v8f a0 = v8f_zero(), a1 = v8f_zero();
      for (int kk = 0; kk < DK; kk += 32) {
        v16bf af = frag_f32(qc, DK, tr * 16, kk, lane);
        v16bf b0 = frag_bf16(kb16, DK, tc0 * 16, kk, lane);
        v16bf b1 = frag_bf16(kb16, DK, (tc0 + 1) * 16, kk, lane);
        a0 = wmma_bf16(af, b0, a0);
        a1 = wmma_bf16(af, b1, a1);
      }
      store_tile(Amat, CHUNK, tr * 16, tc0 * 16, lane, a0, 1.f);
      store_tile(Amat, CHUNK, tr * 16, (tc0 + 1) * 16, lane, a1, 1.f);
    }
    __syncthreads();
    // ---- attn = tril(QK * decay); qc *= e^g -----------------------------
    for (int i = tid; i < CHUNK * CHUNK; i += 256) {
      int r = i >> 6, j = i & 63;
      Amat[i] = (j <= r) ? Amat[i] * __expf(gc[r] - gc[j]) : 0.f;
    }
    for (int j = tid; j < CHUNK * DK / 4; j += 256) {
      int r = j >> 5;
      float4 qv = ((float4*)qc)[j];
      float e = eg[r];
      qv.x *= e; qv.y *= e; qv.z *= e; qv.w *= e;
      ((float4*)qc)[j] = qv;
    }
    __syncthreads();
    // ---- (d) v_new = (-w) @ S + T @ (v*beta) -> bufV --------------------
    {
      int tr = w >> 1, tc0 = (w & 1) * 4;
      v8f a[4];
#pragma unroll
      for (int t = 0; t < 4; ++t) a[t] = v8f_zero();
      for (int kk = 0; kk < DK; kk += 32) {  // (-w) @ S
        v16bf af = frag_bf16(Wb, DK, tr * 16, kk, lane);
#pragma unroll
        for (int t = 0; t < 4; ++t) {
          v16bf b = frag_bf16(St, DK, (tc0 + t) * 16, kk, lane);
          a[t] = wmma_bf16(af, b, a[t]);
        }
      }
      for (int kk = 0; kk < CHUNK; kk += 32) {  // + T @ (v*beta)
        v16bf af = frag_f32(Tm, CHUNK, tr * 16, kk, lane);
#pragma unroll
        for (int t = 0; t < 4; ++t) {
          v16bf b = frag_bf16(Vt, CHUNK, (tc0 + t) * 16, kk, lane);
          a[t] = wmma_bf16(af, b, a[t]);
        }
      }
#pragma unroll
      for (int t = 0; t < 4; ++t)
        store_tile(bufV, DV, tr * 16, (tc0 + t) * 16, lane, a[t], 1.f);
    }
    __syncthreads();
    // ---- Vt = v_new^T bf16 ----------------------------------------------
    for (int j = tid; j < CHUNK * DK / 4; j += 256) {
      int r = j >> 5, d0 = (j & 31) * 4;
      float4 vv = ((const float4*)bufV)[j];
      Vt[(d0 + 0) * CHUNK + r] = (__bf16)vv.x;
      Vt[(d0 + 1) * CHUNK + r] = (__bf16)vv.y;
      Vt[(d0 + 2) * CHUNK + r] = (__bf16)vv.z;
      Vt[(d0 + 3) * CHUNK + r] = (__bf16)vv.w;
    }
    __syncthreads();
    // ---- (e) out = (q*e^g) @ S + attn @ v_new -> global -----------------
    {
      int tr = w >> 1, tc0 = (w & 1) * 4;
      v8f a[4];
#pragma unroll
      for (int t = 0; t < 4; ++t) a[t] = v8f_zero();
      for (int kk = 0; kk < DK; kk += 32) {  // qg @ S
        v16bf af = frag_f32(qc, DK, tr * 16, kk, lane);
#pragma unroll
        for (int t = 0; t < 4; ++t) {
          v16bf b = frag_bf16(St, DK, (tc0 + t) * 16, kk, lane);
          a[t] = wmma_bf16(af, b, a[t]);
        }
      }
      for (int kk = 0; kk < CHUNK; kk += 32) {  // + attn @ v_new
        v16bf af = frag_f32(Amat, CHUNK, tr * 16, kk, lane);
#pragma unroll
        for (int t = 0; t < 4; ++t) {
          v16bf b = frag_bf16(Vt, CHUNK, (tc0 + t) * 16, kk, lane);
          a[t] = wmma_bf16(af, b, a[t]);
        }
      }
      float* ob = coreo + (size_t)s0 * VAL_DIM + (size_t)h * DV;
#pragma unroll
      for (int t = 0; t < 4; ++t)
        store_tile(ob, VAL_DIM, tr * 16, (tc0 + t) * 16, lane, a[t], 1.f);
    }
    __syncthreads();
    // ---- state: S = S*e^gl + (k*e^(gl-g))^T @ v_new ---------------------
    {
      float egl = __expf(glast);
      for (int j = tid; j < DK * DV / 4; j += 256) {
        float4 sv = ((float4*)S)[j];
        sv.x *= egl; sv.y *= egl; sv.z *= egl; sv.w *= egl;
        ((float4*)S)[j] = sv;
      }
    }
    __syncthreads();
    {
      int tr = w;  // wave w: state rows [w*16, +16), all 8 col tiles
      v16bf af0 = frag_bf16(Kd, CHUNK, tr * 16, 0, lane);
      v16bf af1 = frag_bf16(Kd, CHUNK, tr * 16, 32, lane);
#pragma unroll
      for (int tc = 0; tc < 8; ++tc) {
        v8f cfr = load_tile(S, DV, tr * 16, tc * 16, lane);
        v16bf b0 = frag_bf16(Vt, CHUNK, tc * 16, 0, lane);
        v16bf b1 = frag_bf16(Vt, CHUNK, tc * 16, 32, lane);
        cfr = wmma_bf16(af0, b0, cfr);
        cfr = wmma_bf16(af1, b1, cfr);
        store_tile(S, DV, tr * 16, tc * 16, lane, cfr, 1.f);
      }
    }
    __syncthreads();
    // ---- refresh St = S^T bf16 ------------------------------------------
    for (int j = tid; j < DK * DV / 4; j += 256) {
      int k = j >> 5, n0_ = (j & 31) * 4;
      float4 sv = ((const float4*)S)[j];
      St[(n0_ + 0) * DK + k] = (__bf16)sv.x;
      St[(n0_ + 1) * DK + k] = (__bf16)sv.y;
      St[(n0_ + 2) * DK + k] = (__bf16)sv.z;
      St[(n0_ + 3) * DK + k] = (__bf16)sv.w;
    }
    __syncthreads();
  }
}

// ---------------------------------------------------------------------------
// Kernel 5: RMSNorm(core) * norm_w * silu(z) -> gated
// ---------------------------------------------------------------------------
__global__ __launch_bounds__(128) void gate_norm(const float* __restrict__ coreo,
                                                 const float* __restrict__ z,
                                                 const float* __restrict__ normw,
                                                 float* __restrict__ gated) {
  __shared__ float red[128];
  const int s = blockIdx.x, h = blockIdx.y, d = threadIdx.x;
  size_t idx = (size_t)s * VAL_DIM + (size_t)h * DV + d;
  float v = coreo[idx];
  red[d] = v * v;
  __syncthreads();
  for (int off = 64; off > 0; off >>= 1) {
    if (d < off) red[d] += red[d + off];
    __syncthreads();
  }
  float scale = rsqrtf(red[0] * (1.f / 128.f) + 1e-6f);
  float zv = z[idx];
  gated[idx] = v * scale * normw[d] * silu1(zv);
}

// ---------------------------------------------------------------------------
extern "C" void kernel_launch(void* const* d_in, const int* in_sizes, int n_in,
                              void* d_out, int out_size, void* d_ws, size_t ws_size,
                              hipStream_t stream) {
  (void)in_sizes; (void)n_in; (void)out_size; (void)ws_size;
  const float* hidden = (const float*)d_in[0];  // [1,SEQ,HS]
  const float* W_qkv  = (const float*)d_in[1];  // [HS,CONV_DIM]
  const float* W_z    = (const float*)d_in[2];  // [HS,VAL_DIM]
  const float* W_b    = (const float*)d_in[3];  // [HS,NV]
  const float* W_a    = (const float*)d_in[4];  // [HS,NV]
  const float* conv_w = (const float*)d_in[5];  // [CONV_DIM,4]
  const float* dt_b   = (const float*)d_in[6];  // [NV]
  const float* A_log  = (const float*)d_in[7];  // [NV]
  const float* norm_w = (const float*)d_in[8];  // [DV]
  const float* W_out  = (const float*)d_in[9];  // [VAL_DIM,HS]
  float* out = (float*)d_out;                   // [1,SEQ,HS]

  float* ws    = (float*)d_ws;
  float* mixed = ws;                                   // [SEQ,CONV_DIM]
  float* qkvc  = mixed + (size_t)SEQ * CONV_DIM;       // [SEQ,CONV_DIM]
  float* zbuf  = qkvc + (size_t)SEQ * CONV_DIM;        // [SEQ,VAL_DIM]
  float* gbuf  = zbuf + (size_t)SEQ * VAL_DIM;         // [SEQ,NV]
  float* bbuf  = gbuf + (size_t)SEQ * NV;              // [SEQ,NV]
  float* coreo = mixed;   // alias: mixed dead after conv
  float* gated = qkvc;    // alias: qkvc dead after delta_core

  dim3 blk(256);
  dim3 gQKV(CONV_DIM / GBN, SEQ / GBM);
  gemm_f32_bf16wmma<<<gQKV, blk, 0, stream>>>(hidden, W_qkv, mixed, SEQ, CONV_DIM, HS);
  dim3 gZ(VAL_DIM / GBN, SEQ / GBM);
  gemm_f32_bf16wmma<<<gZ, blk, 0, stream>>>(hidden, W_z, zbuf, SEQ, VAL_DIM, HS);
  gb_kernel<<<SEQ, 64, 0, stream>>>(hidden, W_b, W_a, dt_b, A_log, bbuf, gbuf);
  conv_silu<<<8192, 256, 0, stream>>>(mixed, conv_w, qkvc);
  delta_core<<<NV, 256, 0, stream>>>(qkvc, gbuf, bbuf, coreo);
  dim3 gGate(SEQ, NV);
  gate_norm<<<gGate, 128, 0, stream>>>(coreo, zbuf, norm_w, gated);
  dim3 gOut(HS / GBN, SEQ / GBM);
  gemm_f32_bf16wmma<<<gOut, blk, 0, stream>>>(gated, W_out, out, SEQ, HS, VAL_DIM);
}
